// WindowAttention_24026047054179
// MI455X (gfx1250) — matrix operand order repaired and checked by reading, not compiled, not run on hardware
//
#include <hip/hip_runtime.h>

typedef __attribute__((ext_vector_type(16))) _Float16 v16h;
typedef __attribute__((ext_vector_type(8)))  _Float16 v8h;
typedef __attribute__((ext_vector_type(8)))  float    v8f;

#define WSZ   7
#define NTOK  49
#define NPAD  64
#define CDIM  128
#define NH    4
#define HD    32
#define CQKV  384
#define NWIN  64

// ---------------------------------------------------------------------------
// WMMA fragment helpers (wave32, v_wmma_f32_16x16x32_f16)
//
// A (16x32 f16): lane holds row M = lane&15; element e<8  -> K = k0 + 8*half + e
//                                           element e>=8 -> K = k0 + 16 + 8*half + (e-8)
// B (32x16 f16): lane holds col N = n0 + (lane&15); elements e -> K as for A
// C/D (16x16 f32): element r -> M = r + 8*half, N = lane&15
// ---------------------------------------------------------------------------

__device__ __forceinline__ v8f wmma_f16(v16h a, v16h b, v8f c) {
  return __builtin_amdgcn_wmma_f32_16x16x32_f16(false, a, false, b, (short)0, c,
                                                false, false);
}

__device__ __forceinline__ v16h load_A(const _Float16* p, int ld, int row0,
                                       int k0, int lane) {
  const int m  = lane & 15;
  const int hf = lane >> 4;
  const _Float16* r = p + (size_t)(row0 + m) * ld + k0 + 8 * hf;
  v8h lo = *(const v8h*)(r);
  v8h hi = *(const v8h*)(r + 16);
  return __builtin_shufflevector(lo, hi, 0, 1, 2, 3, 4, 5, 6, 7, 8, 9, 10, 11,
                                 12, 13, 14, 15);
}

// A operand where f16 values live in the LOW 16 bits of f32 slots (stride 4B):
// the softmax probabilities P written in-place into the S buffer.
__device__ __forceinline__ v16h load_A_p16of32(const float* Sb, int row0,
                                               int k0, int lane) {
  const int m  = lane & 15;
  const int hf = lane >> 4;
  const float* r = Sb + (size_t)(row0 + m) * NPAD + k0 + 8 * hf;
  v8h lo, hi;
#pragma unroll
  for (int e = 0; e < 8; ++e) lo[e] = *(const _Float16*)(r + e);
#pragma unroll
  for (int e = 0; e < 8; ++e) hi[e] = *(const _Float16*)(r + 16 + e);
  return __builtin_shufflevector(lo, hi, 0, 1, 2, 3, 4, 5, 6, 7, 8, 9, 10, 11,
                                 12, 13, 14, 15);
}

__device__ __forceinline__ v16h load_B(const _Float16* p, int ld, int k0,
                                       int n0, int lane) {
  const int n  = lane & 15;
  const int hf = lane >> 4;
  const _Float16* r = p + (size_t)(k0 + 8 * hf) * ld + n0 + n;
  v16h v;
#pragma unroll
  for (int e = 0; e < 8; ++e) {
    v[e]     = r[(size_t)e * ld];
    v[8 + e] = r[(size_t)(16 + e) * ld];
  }
  return v;
}

// ---------------------------------------------------------------------------
// Prep: weights fp32 -> f16, plus expanded zero-padded bias table
//   biasx[h][i][j] (i,j in [0,64), zero outside 49x49)
// ---------------------------------------------------------------------------
__global__ void wa_prep(const float* __restrict__ wqkv,
                        const float* __restrict__ wproj,
                        const float* __restrict__ btab,
                        _Float16* __restrict__ wqkv_h,
                        _Float16* __restrict__ wproj_h,
                        float* __restrict__ biasx) {
  int i = blockIdx.x * blockDim.x + threadIdx.x;
  if (i < CDIM * CQKV) wqkv_h[i] = (_Float16)wqkv[i];
  if (i < CDIM * CDIM) wproj_h[i] = (_Float16)wproj[i];
  if (i < NH * NPAD * NPAD) {
    int h = i >> 12, rem = i & 4095;
    int ii = rem >> 6, jj = rem & 63;
    float val = 0.0f;
    if (ii < NTOK && jj < NTOK) {
      int ri = ii / WSZ, ci = ii % WSZ, rj = jj / WSZ, cj = jj % WSZ;
      int idx = (ri - rj + WSZ - 1) * (2 * WSZ - 1) + (ci - cj + WSZ - 1);
      val = btab[idx * NH + h];
    }
    biasx[i] = val;
  }
}

// ---------------------------------------------------------------------------
// Fused window attention: one block (128 threads = 4 waves) per window.
//
// LDS map (bytes):
//   [0,     16384) xs  f16[64][128]  -- input tile; reused as O f16[64][128]
//   [16384, 32768) q   f16[4][64][32]  (pre-scaled by hd^-0.5)
//   [32768, 49152) kT  f16[4][32][64]  (transposed = B-operand layout)
//   [49152, 65536) v   f16[4][64][32]
//   [65536, 81920) S   f32[64][64]     (logits; P f16 written in-place)
//   [81920, 82176) rsum f32[64]
// ---------------------------------------------------------------------------
__global__ __launch_bounds__(128) void wa_fused_kernel(
    const float* __restrict__ x, const float* __restrict__ mask,
    const _Float16* __restrict__ wqkv_h, const float* __restrict__ b_qkv,
    const _Float16* __restrict__ wproj_h, const float* __restrict__ b_proj,
    const float* __restrict__ biasx, float* __restrict__ out) {
  __shared__ alignas(32) char smem[82176];
  _Float16* xs = (_Float16*)(smem);            // [64][128], later O
  _Float16* qs = (_Float16*)(smem + 16384);    // [4][64][32]
  _Float16* kT = (_Float16*)(smem + 32768);    // [4][32][64]
  _Float16* vs = (_Float16*)(smem + 49152);    // [4][64][32]
  float*    S  = (float*)(smem + 65536);       // [64][64]
  float*    rs = (float*)(smem + 81920);       // [64]

  const int b    = blockIdx.x;
  const int tid  = threadIdx.x;
  const int lane = tid & 31;
  const int wv   = tid >> 5;
  const int tn   = lane & 15;
  const int hf   = lane >> 4;
  const float scale = 0.17677669529663687f;  // 32^-0.5

  // ---- load x tile, fp32 -> f16, zero-pad rows [49,64) -----------------
  const float* xg = x + (size_t)b * NTOK * CDIM;
  for (int i = tid; i < NPAD * CDIM; i += 128) {
    int r = i >> 7, c = i & 127;
    float val = (r < NTOK) ? xg[r * CDIM + c] : 0.0f;
    xs[i] = (_Float16)val;
  }
  __syncthreads();

  // ---- Stage 1: qkv = xs @ w_qkv + b_qkv; scatter to q / kT / v --------
  // Per N-tile: weight B-fragments in registers; kt-outer / mt-inner with
  // 4 live accumulators -> each bfr[kt] feeds 4 back-to-back independent
  // WMMAs (structural reuse + hidden XDL RAW latency).
  for (int nt = wv; nt < 24; nt += 4) {
    v16h bfr[4];
#pragma unroll
    for (int kt = 0; kt < 4; ++kt)
      bfr[kt] = load_B(wqkv_h, CQKV, kt * 32, nt * 16, lane);

    v8f acc[4];
#pragma unroll
    for (int mt = 0; mt < 4; ++mt)
      acc[mt] = (v8f){0.f, 0.f, 0.f, 0.f, 0.f, 0.f, 0.f, 0.f};
#pragma unroll
    for (int kt = 0; kt < 4; ++kt)
#pragma unroll
      for (int mt = 0; mt < 4; ++mt)
        acc[mt] =
            wmma_f16(load_A(xs, CDIM, mt * 16, kt * 32, lane), bfr[kt], acc[mt]);

    const int col  = nt * 16 + tn;
    const float bq = b_qkv[col];
    // nt is wave-uniform (seeded by tid>>5); force scatter selectors into
    // SGPRs so the 3-way branch is scalar, not EXEC-masked.
    const int which = __builtin_amdgcn_readfirstlane(nt >> 3);       // 0=q,1=k,2=v
    const int head  = __builtin_amdgcn_readfirstlane((nt >> 1) & 3); // head id
    const int dd    = ((nt & 1) << 4) + tn;

    _Float16* qdst = qs + head * (NPAD * HD) + dd;        // + m*HD
    _Float16* kdst = kT + head * (HD * NPAD) + dd * NPAD; // + m
    _Float16* vdst = vs + head * (NPAD * HD) + dd;        // + m*HD

    if (which == 0) {
#pragma unroll
      for (int mt = 0; mt < 4; ++mt)
#pragma unroll
        for (int r = 0; r < 8; ++r) {
          int m = mt * 16 + r + 8 * hf;
          qdst[m * HD] = (_Float16)((acc[mt][r] + bq) * scale);
        }
    } else if (which == 1) {
#pragma unroll
      for (int mt = 0; mt < 4; ++mt)
#pragma unroll
        for (int r = 0; r < 8; ++r) {
          int m = mt * 16 + r + 8 * hf;
          kdst[m] = (_Float16)(acc[mt][r] + bq);  // transposed store
        }
    } else {
#pragma unroll
      for (int mt = 0; mt < 4; ++mt)
#pragma unroll
        for (int r = 0; r < 8; ++r) {
          int m = mt * 16 + r + 8 * hf;
          vdst[m * HD] = (_Float16)(acc[mt][r] + bq);
        }
    }
  }
  __syncthreads();

  // ---- Stages 2+3 per head --------------------------------------------
  const int wIdx = b & (NWIN - 1);
  const float* maskw = mask + (size_t)wIdx * NTOK * NTOK;

  for (int h = 0; h < NH; ++h) {
    const _Float16* qh = qs + h * NPAD * HD;
    const _Float16* kh = kT + h * HD * NPAD;
    const _Float16* vh = vs + h * NPAD * HD;
    const float*    bx = biasx + h * NPAD * NPAD;

    // S = (q*scale) @ k^T : wave wv owns N-tile wv; B-fragment hoisted.
    {
      const v16h bm = load_B(kh, NPAD, 0, wv * 16, lane);
      const int j  = wv * 16 + tn;
      const int jm = (j < NTOK) ? j : (NTOK - 1);
      v8f sacc[4];
#pragma unroll
      for (int mt = 0; mt < 4; ++mt) {
        v16h a = load_A(qh, HD, mt * 16, 0, lane);
        v8f z  = {0.f, 0.f, 0.f, 0.f, 0.f, 0.f, 0.f, 0.f};
        sacc[mt] = wmma_f16(a, bm, z);
      }
#pragma unroll
      for (int mt = 0; mt < 4; ++mt)
#pragma unroll
        for (int r = 0; r < 8; ++r) {
          int i  = mt * 16 + r + 8 * hf;
          int im = (i < NTOK) ? i : (NTOK - 1);
          // padded cols of S are never read by softmax; padded rows only feed unstored rows -> don't-cares
          S[i * NPAD + j] = sacc[mt][r] + bx[i * NPAD + j] + maskw[im * NTOK + jm];
        }
    }
    __syncthreads();

    // Row softmax; write P f16 in-place into low half of each f32 slot.
    if (tid < NPAD) {
      float* Sr = S + tid * NPAD;
      float mx = -1e30f;
      for (int j = 0; j < NTOK; ++j) mx = fmaxf(mx, Sr[j]);
      float sum = 0.f;
      for (int j = 0; j < NTOK; ++j) {
        float p = __expf(Sr[j] - mx);
        sum += p;
        *(_Float16*)(Sr + j) = (_Float16)p;
      }
      for (int j = NTOK; j < NPAD; ++j) *(_Float16*)(Sr + j) = (_Float16)0.0f;
      rs[tid] = 1.0f / sum;
    }
    __syncthreads();

    // O_h = (P @ v_h) * (1/rowsum): wave wv owns N-tile (wv&1),
    // M-tiles (wv>>1) and (wv>>1)+2; V B-fragments hoisted; kt-outer with
    // 2 live accumulators.
    {
      const int nt = wv & 1;
      v16h bv[2];
      bv[0] = load_B(vh, HD, 0, nt * 16, lane);
      bv[1] = load_B(vh, HD, 32, nt * 16, lane);
      const int col = h * HD + nt * 16 + tn;
      v8f oacc[2];
#pragma unroll
      for (int t = 0; t < 2; ++t)
        oacc[t] = (v8f){0.f, 0.f, 0.f, 0.f, 0.f, 0.f, 0.f, 0.f};
#pragma unroll
      for (int kt = 0; kt < 2; ++kt)
#pragma unroll
        for (int t = 0; t < 2; ++t) {
          int mt = (wv >> 1) + 2 * t;
          oacc[t] =
              wmma_f16(load_A_p16of32(S, mt * 16, kt * 32, lane), bv[kt], oacc[t]);
        }
#pragma unroll
      for (int t = 0; t < 2; ++t) {
        int mt = (wv >> 1) + 2 * t;
#pragma unroll
        for (int r = 0; r < 8; ++r) {
          int i = mt * 16 + r + 8 * hf;
          xs[i * CDIM + col] = (_Float16)(oacc[t][r] * rs[i]);  // O region
        }
      }
    }
    __syncthreads();
  }

  // ---- Stage 4: out = O @ w_proj + b_proj ------------------------------
  // N-tile outer per wave; w_proj B-fragments held in registers; kt-outer /
  // mt-inner with 4 live accumulators.
  float* og = out + (size_t)b * NTOK * CDIM;
  for (int nt = wv; nt < 8; nt += 4) {
    v16h bfr[4];
#pragma unroll
    for (int kt = 0; kt < 4; ++kt)
      bfr[kt] = load_B(wproj_h, CDIM, kt * 32, nt * 16, lane);

    v8f acc[4];
#pragma unroll
    for (int mt = 0; mt < 4; ++mt)
      acc[mt] = (v8f){0.f, 0.f, 0.f, 0.f, 0.f, 0.f, 0.f, 0.f};
#pragma unroll
    for (int kt = 0; kt < 4; ++kt)
#pragma unroll
      for (int mt = 0; mt < 4; ++mt)
        acc[mt] =
            wmma_f16(load_A(xs, CDIM, mt * 16, kt * 32, lane), bfr[kt], acc[mt]);

    const int col  = nt * 16 + tn;
    const float bp = b_proj[col];
#pragma unroll
    for (int mt = 0; mt < 4; ++mt)
#pragma unroll
      for (int r = 0; r < 8; ++r) {
        int i = mt * 16 + r + 8 * hf;
        if (i < NTOK) og[i * CDIM + col] = acc[mt][r] + bp;
      }
  }
}

// ---------------------------------------------------------------------------
extern "C" void kernel_launch(void* const* d_in, const int* in_sizes, int n_in,
                              void* d_out, int out_size, void* d_ws,
                              size_t ws_size, hipStream_t stream) {
  (void)in_sizes; (void)n_in; (void)out_size; (void)ws_size;
  const float* x     = (const float*)d_in[0];
  const float* mask  = (const float*)d_in[1];
  const float* wqkv  = (const float*)d_in[2];
  const float* bqkv  = (const float*)d_in[3];
  const float* wproj = (const float*)d_in[4];
  const float* bproj = (const float*)d_in[5];
  const float* btab  = (const float*)d_in[6];

  // workspace layout: wqkv f16 (98304B) | wproj f16 (32768B) | biasx f32 (65536B)
  _Float16* wqkv_h  = (_Float16*)d_ws;
  _Float16* wproj_h = wqkv_h + CDIM * CQKV;
  float*    biasx   = (float*)((char*)d_ws + 131072);

  wa_prep<<<192, 256, 0, stream>>>(wqkv, wproj, btab, wqkv_h, wproj_h, biasx);
  wa_fused_kernel<<<4096, 128, 0, stream>>>(x, mask, wqkv_h, bqkv, wproj_h,
                                            bproj, biasx, (float*)d_out);
}
